// GE2ELoss_47648367182064
// MI455X (gfx1250) — matrix operand order repaired and checked by reading, not compiled, not run on hardware
//
#include <hip/hip_runtime.h>

typedef __attribute__((ext_vector_type(16))) _Float16 v16h;
typedef __attribute__((ext_vector_type(8)))  float    v8f;
typedef __attribute__((ext_vector_type(4)))  _Float16 h4;
typedef __attribute__((ext_vector_type(4)))  unsigned int u32x4;
typedef __attribute__((ext_vector_type(8)))  int i32x8;
typedef __attribute__((ext_vector_type(4)))  int i32x4;

#define N_SPK 1024
#define M_UTT 32
#define D_EMB 512
#define NM_ROWS (N_SPK * M_UTT)

#define ROWS_BLK 128
#define COLS_BLK 128
#define K_STEPS (D_EMB / 32)            // 16 WMMA K-steps
#define ROW_BLOCKS (NM_ROWS / ROWS_BLK) // 256
#define COL_BLOCKS (N_SPK / COLS_BLK)   // 8
#define N_CGRP (N_SPK / 16)             // 64 column groups of 16
#define ROW_RED_BLOCKS (NM_ROWS / 256)  // 128

// ---------------------------------------------------------------------------
// Kernel 1: per-speaker centroids (f16), f16 copy of embeddings (for TDM
// staging in kernel 2), and exact f32 leave-one-out self-similarity.
// ---------------------------------------------------------------------------
__global__ __launch_bounds__(256) void centroid_kernel(
    const float* __restrict__ e,
    _Float16* __restrict__ c16,
    _Float16* __restrict__ e16,
    float* __restrict__ self_sim) {
  const int n = blockIdx.x;
  const int tid = threadIdx.x;
  __shared__ float sS[D_EMB];

  const float* en = e + (size_t)n * M_UTT * D_EMB;
  _Float16* e16n = e16 + (size_t)n * M_UTT * D_EMB;

  for (int d = tid; d < D_EMB; d += 256) {
    float acc = 0.f;
#pragma unroll
    for (int m = 0; m < M_UTT; ++m) {
      float v = en[m * D_EMB + d];
      e16n[m * D_EMB + d] = (_Float16)v;
      acc += v;
    }
    sS[d] = acc;
    c16[(size_t)n * D_EMB + d] = (_Float16)(acc * (1.0f / M_UTT));
  }
  __syncthreads();

  const int wave = tid >> 5;
  const int lane = tid & 31;
  for (int mm = wave; mm < M_UTT; mm += 8) {
    float se = 0.f, ee = 0.f;
    const float* em = en + mm * D_EMB;
    for (int d = lane; d < D_EMB; d += 32) {
      float v = em[d];
      se += sS[d] * v;
      ee += v * v;
    }
#pragma unroll
    for (int off = 16; off; off >>= 1) {
      se += __shfl_xor(se, off, 32);
      ee += __shfl_xor(ee, off, 32);
    }
    if (lane == 0)
      self_sim[n * M_UTT + mm] = (se - ee) * (1.0f / (M_UTT - 1));
  }
}

// ---------------------------------------------------------------------------
// Kernel 2: WMMA GEMM. A tile (128 x 512 f16 = 128 KB) staged into LDS by the
// Tensor Data Mover (one descriptor, issued by wave 0, overlapped with the
// global B-fragment loads). B register-resident (16 cols x 512 K = 128 VGPRs
// per wave). A fragments double-buffered from LDS so the DS fetch of step k+1
// overlaps the WMMA of step k. Emits per-(row, col-group) softmax partials.
// ---------------------------------------------------------------------------
__global__ __launch_bounds__(256) void gemm_stats_kernel(
    const _Float16* __restrict__ e16,
    const _Float16* __restrict__ c16,
    const float* __restrict__ self_sim,
    const float* __restrict__ wp,
    const float* __restrict__ bp,
    float* __restrict__ stats,   // [NM_ROWS][N_CGRP][2]
    float* __restrict__ llab) {  // [NM_ROWS]
  extern __shared__ _Float16 ldsA[];        // ROWS_BLK * D_EMB halves = 128 KB

  const int tid = threadIdx.x;
  const int rb = blockIdx.x * ROWS_BLK;
  const int cb = blockIdx.y * COLS_BLK;
  const float w = wp[0];
  const float b = bp[0];

  const int wave = tid >> 5;
  const int lane = tid & 31;
  const int hl = lane >> 4;                 // half-wave select
  const int l15 = lane & 15;
  const int col = cb + wave * 16 + l15;     // this lane's output column
  const int cbg = blockIdx.y * 8 + wave;    // global column group [0,64)

  // ---- wave 0: issue TDM copy of A tile (rows rb..rb+127) into LDS -----
  if (wave == 0) {
    unsigned long long ga =
        (unsigned long long)(uintptr_t)(e16 + (size_t)rb * D_EMB);
    unsigned int lds_off = (unsigned int)(uintptr_t)ldsA;  // low 32b = LDS byte offset

    u32x4 g0;
    g0[0] = 1u;                                   // count=1, user descriptor
    g0[1] = lds_off;                              // lds_addr (bytes)
    g0[2] = (unsigned int)ga;                     // global_addr[31:0]
    g0[3] = (unsigned int)((ga >> 32) & 0x1FFFFFFu) | (2u << 30);  // [56:32]|type=2

    i32x8 g1;
    g1[0] = 1 << 16;                              // data_size=2B, mask=0
    g1[1] = (int)((unsigned)D_EMB << 16);         // tensor_dim0[15:0]
    g1[2] = (int)((unsigned)ROWS_BLK << 16);      // tensor_dim0[31:16]=0 | tensor_dim1[15:0]
    g1[3] = (int)((unsigned)D_EMB << 16);         // tensor_dim1[31:16]=0 | tile_dim0
    g1[4] = ROWS_BLK;                             // tile_dim1 (tile_dim2=0)
    g1[5] = D_EMB;                                // tensor_dim0_stride[31:0]
    g1[6] = 0;                                    // stride0[47:32] | stride1[15:0]
    g1[7] = 0;                                    // stride1[47:16]

    i32x4 z4 = {0, 0, 0, 0};
    i32x8 z8 = {0, 0, 0, 0, 0, 0, 0, 0};
    __builtin_amdgcn_tensor_load_to_lds(g0, g1, z4, z4, z8, 0);
  }

  union V16 { uint4 q[2]; v16h h; };

  // ---- all waves: load B fragments (overlaps with the TDM DMA) ---------
  // Contiguous k order (A matches): lanes<16 hold K 0-15 of col, lanes>=16 hold K 16-31.
  V16 breg[K_STEPS];
  {
    const uint4* bq = (const uint4*)(c16 + (size_t)col * D_EMB) + hl * 2;
#pragma unroll
    for (int k = 0; k < K_STEPS; ++k) {
      breg[k].q[0] = bq[k * 4];
      breg[k].q[1] = bq[k * 4 + 1];
    }
  }

  if (wave == 0) __builtin_amdgcn_s_wait_tensorcnt(0);
  __syncthreads();

  const uint4* aq = (const uint4*)ldsA;

  // ---- sweep the 8 row tiles, streaming A fragments from LDS -----------
  for (int t = 0; t < ROWS_BLK / 16; ++t) {
    // A layout (matches B): lanes<16 hold K 0-15, lanes>=16 hold K 16-31
    const int abase = (t * 16 + l15) * (D_EMB / 8) + hl * 2;
    v8f acc = {};

    V16 a0;
    a0.q[0] = aq[abase];
    a0.q[1] = aq[abase + 1];
#pragma unroll
    for (int k = 0; k < K_STEPS; ++k) {
      V16 an = a0;
      if (k + 1 < K_STEPS) {                      // prefetch next A fragment
        an.q[0] = aq[abase + (k + 1) * 4];
        an.q[1] = aq[abase + (k + 1) * 4 + 1];
      }
      acc = __builtin_amdgcn_wmma_f32_16x16x32_f16(
          false, a0.h, false, breg[k].h, (short)0, acc, false, false);
      a0 = an;
    }

    // per-row (max, sumexp) over this wave's 16 columns
#pragma unroll
    for (int g = 0; g < 8; ++g) {
      const int row = rb + t * 16 + g + hl * 8;   // global row
      float v = acc[g];
      float logit = w * v + b;
      if (col == (row >> 5)) {                    // label column for this row
        v = self_sim[row];                        // exact leave-one-out value
        logit = w * v + b;
        llab[row] = logit;
      }
      float m = logit;
#pragma unroll
      for (int off = 1; off < 16; off <<= 1)
        m = fmaxf(m, __shfl_xor(m, off, 32));
      float sx = __expf(logit - m);
#pragma unroll
      for (int off = 1; off < 16; off <<= 1)
        sx += __shfl_xor(sx, off, 32);
      if (l15 == 0) {
        float* sp = stats + ((size_t)row * N_CGRP + cbg) * 2;
        sp[0] = m;
        sp[1] = sx;
      }
    }
  }
}

// ---------------------------------------------------------------------------
// Kernel 3: combine 64 (max,sumexp) partials per row -> row loss; reduce
// 256 rows per block into a block partial.
// ---------------------------------------------------------------------------
__global__ __launch_bounds__(256) void rowloss_kernel(
    const float* __restrict__ stats,
    const float* __restrict__ llab,
    float* __restrict__ partials) {
  const int row = blockIdx.x * 256 + threadIdx.x;
  const float* sp = stats + (size_t)row * N_CGRP * 2;

  float m = -__builtin_huge_valf();
#pragma unroll 8
  for (int j = 0; j < N_CGRP; ++j) m = fmaxf(m, sp[2 * j]);
  float s = 0.f;
#pragma unroll 8
  for (int j = 0; j < N_CGRP; ++j) s += sp[2 * j + 1] * __expf(sp[2 * j] - m);

  float loss = m + __logf(s) - llab[row];

  __shared__ float sm[256];
  sm[threadIdx.x] = loss;
  __syncthreads();
  for (int st = 128; st; st >>= 1) {
    if (threadIdx.x < st) sm[threadIdx.x] += sm[threadIdx.x + st];
    __syncthreads();
  }
  if (threadIdx.x == 0) partials[blockIdx.x] = sm[0];
}

// ---------------------------------------------------------------------------
// Kernel 4: deterministic reduction of block partials -> mean loss
// ---------------------------------------------------------------------------
__global__ __launch_bounds__(256) void finalize_kernel(
    const float* __restrict__ partials, float* __restrict__ out) {
  __shared__ float sm[256];
  float acc = 0.f;
  for (int i = threadIdx.x; i < ROW_RED_BLOCKS; i += 256) acc += partials[i];
  sm[threadIdx.x] = acc;
  __syncthreads();
  for (int s = 128; s; s >>= 1) {
    if (threadIdx.x < s) sm[threadIdx.x] += sm[threadIdx.x + s];
    __syncthreads();
  }
  if (threadIdx.x == 0) out[0] = sm[0] * (1.0f / NM_ROWS);
}

extern "C" void kernel_launch(void* const* d_in, const int* in_sizes, int n_in,
                              void* d_out, int out_size, void* d_ws,
                              size_t ws_size, hipStream_t stream) {
  const float* e = (const float*)d_in[0];
  const float* w = (const float*)d_in[1];
  const float* b = (const float*)d_in[2];
  (void)in_sizes; (void)n_in; (void)out_size; (void)ws_size;

  char* ws = (char*)d_ws;
  size_t off = 0;
  _Float16* c16 = (_Float16*)(ws + off);
  off += (size_t)N_SPK * D_EMB * sizeof(_Float16);            // 1 MB
  _Float16* e16 = (_Float16*)(ws + off);
  off += (size_t)NM_ROWS * D_EMB * sizeof(_Float16);          // 32 MB
  float* self_sim = (float*)(ws + off);
  off += (size_t)NM_ROWS * sizeof(float);                     // 128 KB
  float* stats = (float*)(ws + off);
  off += (size_t)NM_ROWS * N_CGRP * 2 * sizeof(float);        // 16 MB
  float* llab = (float*)(ws + off);
  off += (size_t)NM_ROWS * sizeof(float);                     // 128 KB
  float* partials = (float*)(ws + off);                       // 512 B

  centroid_kernel<<<N_SPK, 256, 0, stream>>>(e, c16, e16, self_sim);

  dim3 grid(ROW_BLOCKS, COL_BLOCKS);
  size_t ldsA_bytes = (size_t)ROWS_BLK * D_EMB * sizeof(_Float16);  // 128 KB
  gemm_stats_kernel<<<grid, 256, ldsA_bytes, stream>>>(
      e16, c16, self_sim, w, b, stats, llab);

  rowloss_kernel<<<ROW_RED_BLOCKS, 256, 0, stream>>>(stats, llab, partials);
  finalize_kernel<<<1, 256, 0, stream>>>(partials, (float*)d_out);
}
